// GRU_59528246722861
// MI455X (gfx1250) — matrix operand order repaired and checked by reading, not compiled, not run on hardware
//
#include <hip/hip_runtime.h>
#include <hip/hip_fp16.h>

// GRU (reset gate unused): S=2048, B=64, I=H=512. Only z/n gates needed (2H cols).
#define S_LEN 2048
#define BATCH 64
#define IDIM  512
#define HDIM  512
#define GDIM  1024  // 2 gates (z, n) * HDIM

typedef _Float16 v16h __attribute__((ext_vector_type(16)));
typedef _Float16 v8h  __attribute__((ext_vector_type(8)));
typedef float    v8f  __attribute__((ext_vector_type(8)));

static __device__ __forceinline__ v16h make_afrag(v8h lo, v8h hi) {
  return __builtin_shufflevector(lo, hi, 0,1,2,3,4,5,6,7,8,9,10,11,12,13,14,15);
}

static __device__ __forceinline__ v8f wmma_f16(v16h a, v16h b, v8f c) {
  // D = A(16x32 f16) x B(32x16 f16) + C(16x16 f32)
  return __builtin_amdgcn_wmma_f32_16x16x32_f16(false, a, false, b, (short)0, c,
                                                false, false);
}

// ---------------------------------------------------------------------------
// Kernel 0: convert z/n-gate rows of W_ih, W_hh to f16; fold biases.
// ---------------------------------------------------------------------------
__global__ void gru_prep(const float* __restrict__ wih, const float* __restrict__ whh,
                         const float* __restrict__ bih, const float* __restrict__ bhh,
                         _Float16* __restrict__ wih16, _Float16* __restrict__ whh16,
                         float* __restrict__ biaszn) {
  const int n = GDIM * IDIM;  // 524288 (z/n rows only: rows H..3H of the 3H x 512)
  for (int i = blockIdx.x * blockDim.x + threadIdx.x; i < n;
       i += gridDim.x * blockDim.x) {
    wih16[i] = (_Float16)wih[HDIM * IDIM + i];
    whh16[i] = (_Float16)whh[HDIM * HDIM + i];
  }
  const int t = blockIdx.x * blockDim.x + threadIdx.x;
  if (t < GDIM) biaszn[t] = bih[HDIM + t] + bhh[HDIM + t];
}

// ---------------------------------------------------------------------------
// Kernel 1: gx[s*B+b, g] = x[s,b,:] . W_ih[H+g,:] + bias  (M=131072,K=512,N=1024)
// One wave -> one 16-row M tile x four 16-col N tiles (A fragment reused 4x).
// ---------------------------------------------------------------------------
__global__ void __launch_bounds__(256) gru_gx(const float* __restrict__ x,
                                              const _Float16* __restrict__ wih16,
                                              const float* __restrict__ biaszn,
                                              float* __restrict__ gx) {
  const int lane  = threadIdx.x & 31;
  const int wid   = (blockIdx.x * blockDim.x + threadIdx.x) >> 5;
  const int mtile = wid >> 4;   // 0..8191 -> rows [mtile*16, +16) of S*B
  const int ngrp  = wid & 15;   // 0..15   -> cols [ngrp*64, +64)
  const int mloc  = lane & 15;
  const int khalf = lane >> 4;
  const int nloc  = lane & 15;
  const int col0  = ngrp * 64;
  const int row0  = mtile * 16;

  const float* arow = x + (size_t)(row0 + mloc) * IDIM;
  // B fragment: lane holds output column (= row of wih16), K contiguous.
  const _Float16* b0 = wih16 + (size_t)(col0 +  0 + nloc) * IDIM + khalf * 16;
  const _Float16* b1 = wih16 + (size_t)(col0 + 16 + nloc) * IDIM + khalf * 16;
  const _Float16* b2 = wih16 + (size_t)(col0 + 32 + nloc) * IDIM + khalf * 16;
  const _Float16* b3 = wih16 + (size_t)(col0 + 48 + nloc) * IDIM + khalf * 16;

  v8f acc0 = {}, acc1 = {}, acc2 = {}, acc3 = {};
  #pragma unroll 4
  for (int k0 = 0; k0 < IDIM; k0 += 32) {
    v8f alo = *(const v8f*)(arow + k0 + khalf * 16);
    v8f ahi = *(const v8f*)(arow + k0 + khalf * 16 + 8);
    v16h af = make_afrag(__builtin_convertvector(alo, v8h),
                         __builtin_convertvector(ahi, v8h));
    v16h f0 = *(const v16h*)(b0 + k0);
    v16h f1 = *(const v16h*)(b1 + k0);
    v16h f2 = *(const v16h*)(b2 + k0);
    v16h f3 = *(const v16h*)(b3 + k0);
    acc0 = wmma_f16(af, f0, acc0);
    acc1 = wmma_f16(af, f1, acc1);
    acc2 = wmma_f16(af, f2, acc2);
    acc3 = wmma_f16(af, f3, acc3);
  }

  // C/D layout: lane L, vgpr v -> (m = v + 8*(L>>4), n = L&15)
  #pragma unroll
  for (int t = 0; t < 4; ++t) {
    const v8f acc = (t == 0) ? acc0 : (t == 1) ? acc1 : (t == 2) ? acc2 : acc3;
    const int col = col0 + t * 16 + nloc;
    const float bb = biaszn[col];
    #pragma unroll
    for (int v = 0; v < 8; ++v)
      gx[(size_t)(row0 + v + 8 * khalf) * GDIM + col] = acc[v] + bb;
  }
}

// ---------------------------------------------------------------------------
// Grid barrier: monotone phase-counting spin on an L2 atomic.
// ---------------------------------------------------------------------------
static __device__ __forceinline__ void grid_barrier(unsigned* bar, int nwg,
                                                    unsigned phase) {
  __syncthreads();
  if (threadIdx.x == 0) {
    __threadfence();
    __hip_atomic_fetch_add(bar, 1u, __ATOMIC_RELEASE, __HIP_MEMORY_SCOPE_AGENT);
    const unsigned target = (unsigned)nwg * phase;
    while (__hip_atomic_load(bar, __ATOMIC_ACQUIRE, __HIP_MEMORY_SCOPE_AGENT) <
           target)
      __builtin_amdgcn_s_sleep(2);
  }
  __syncthreads();
  __threadfence();
}

// ---------------------------------------------------------------------------
// Kernel 2: persistent recurrence. 16 WGs x 8 waves = 128 waves; each wave owns
// one 16x16 z-tile and its paired n-tile (shared A = h in f16). One barrier per
// step. W_hh B-fragments are hoisted into 256 VGPRs before the 2048-step loop;
// gx/h_prev loads are issued before the WMMA chain; K split into even/odd
// accumulator chains so the XDL pipe stays full.
// ---------------------------------------------------------------------------
__global__ void __launch_bounds__(256, 1) gru_recur(
    const float* __restrict__ gx, const float* __restrict__ h0,
    const _Float16* __restrict__ whh16, float* __restrict__ out,
    float* __restrict__ hn_out, float* __restrict__ hbuf,
    _Float16* __restrict__ h16buf, unsigned* __restrict__ bar, int nwg) {
  const int lane  = threadIdx.x & 31;
  const int wid   = blockIdx.x * (blockDim.x >> 5) + (threadIdx.x >> 5);  // 0..127
  const int mt    = wid >> 5;   // 0..3   -> batch rows [mt*16, +16)
  const int ntz   = wid & 31;   // 0..31  -> hidden cols [ntz*16, +16)
  const int mloc  = lane & 15;
  const int khalf = lane >> 4;
  const int nloc  = lane & 15;
  const int j     = ntz * 16 + nloc;  // hidden column owned by this lane

  // Initialize both h buffers from h_0 (4096 threads x 8 elems = 32768).
  {
    const int tid = blockIdx.x * blockDim.x + threadIdx.x;
    #pragma unroll
    for (int e = 0; e < 8; ++e) {
      const int i = tid * 8 + e;
      const float v = h0[i];
      hbuf[i] = v;
      h16buf[i] = (_Float16)v;
    }
  }
  grid_barrier(bar, nwg, 1);

  // Preload ALL W_hh B-fragments for both gates: invariant across all steps.
  // z gate uses whh16 rows [0,512), n gate rows [512,1024). 32 x v16h = 256 VGPRs.
  const _Float16* bz = whh16 + (size_t)(ntz * 16 + nloc) * HDIM + khalf * 16;
  const _Float16* bn = whh16 + (size_t)(512 + ntz * 16 + nloc) * HDIM + khalf * 16;
  v16h BZ[16], BN[16];
  #pragma unroll
  for (int kk = 0; kk < 16; ++kk) {
    BZ[kk] = *(const v16h*)(bz + kk * 32);
    BN[kk] = *(const v16h*)(bn + kk * 32);
  }

  const _Float16* hA0 = h16buf + (size_t)(mt * 16 + mloc) * HDIM;  // buffer 0
  const _Float16* hA1 = hA0 + BATCH * HDIM;                        // buffer 1

  for (int s = 0; s < S_LEN; ++s) {
    const int cur = s & 1, nxt = cur ^ 1;
    const float* gxs = gx + (size_t)s * (BATCH * GDIM);
    const float* hc  = hbuf + cur * (BATCH * HDIM);

    // Issue the loads that do NOT depend on this step's GEMM first, so they
    // overlap the WMMA chain: gate pre-activations (HBM stream) + h_prev f32.
    float gzv[8], gnv[8], hpv[8];
    #pragma unroll
    for (int v = 0; v < 8; ++v) {
      const int b = mt * 16 + v + 8 * khalf;
      gzv[v] = gxs[(size_t)b * GDIM + j];
      gnv[v] = gxs[(size_t)b * GDIM + 512 + j];
      hpv[v] = hc[(size_t)b * HDIM + j];
    }

    // h @ W_hh^T for the z/n tiles. Even/odd K accumulator chains (4 chains of
    // 8 dependent WMMAs instead of 2 chains of 16).
    const _Float16* hA = cur ? hA1 : hA0;
    v8f az0 = {}, az1 = {}, an0 = {}, an1 = {};
    #pragma unroll
    for (int kk = 0; kk < 16; kk += 2) {
      const int k0 = kk * 32, k1 = k0 + 32;
      v16h af0 = make_afrag(*(const v8h*)(hA + k0 + khalf * 16),
                            *(const v8h*)(hA + k0 + khalf * 16 + 8));
      v16h af1 = make_afrag(*(const v8h*)(hA + k1 + khalf * 16),
                            *(const v8h*)(hA + k1 + khalf * 16 + 8));
      az0 = wmma_f16(af0, BZ[kk], az0);
      an0 = wmma_f16(af0, BN[kk], an0);
      az1 = wmma_f16(af1, BZ[kk + 1], az1);
      an1 = wmma_f16(af1, BN[kk + 1], an1);
    }
    const v8f accz = az0 + az1;
    const v8f accn = an0 + an1;

    float*    hx   = hbuf + nxt * (BATCH * HDIM);
    _Float16* hx16 = h16buf + nxt * (BATCH * HDIM);

    #pragma unroll
    for (int v = 0; v < 8; ++v) {
      const int b = mt * 16 + v + 8 * khalf;
      const float gz = gzv[v] + accz[v];
      const float gn = gnv[v] + accn[v];
      const float zg = 1.0f / (1.0f + __expf(-gz));  // sigmoid
      const float ng = tanhf(gn);
      const float hv = fmaf(zg, ng - hpv[v], hpv[v]);  // (1-z)*h + z*n
      out[((size_t)s * BATCH + b) * HDIM + j] = hv;
      hx[(size_t)b * HDIM + j] = hv;
      hx16[(size_t)b * HDIM + j] = (_Float16)hv;
      if (s + 1 < S_LEN) {  // prefetch next step's gx rows (streams from HBM)
        __builtin_prefetch(&gxs[(size_t)(BATCH * GDIM) + (size_t)b * GDIM + j], 0, 1);
        __builtin_prefetch(&gxs[(size_t)(BATCH * GDIM) + (size_t)b * GDIM + 512 + j], 0, 1);
      }
    }
    if (s == S_LEN - 1) {
      #pragma unroll
      for (int v = 0; v < 8; ++v) {
        const int b = mt * 16 + v + 8 * khalf;
        hn_out[(size_t)b * HDIM + j] = hx[(size_t)b * HDIM + j];
      }
    }
    grid_barrier(bar, nwg, (unsigned)(s + 2));
  }
}

// ---------------------------------------------------------------------------
extern "C" void kernel_launch(void* const* d_in, const int* in_sizes, int n_in,
                              void* d_out, int out_size, void* d_ws, size_t ws_size,
                              hipStream_t stream) {
  const float* x   = (const float*)d_in[0];
  const float* h0  = (const float*)d_in[1];
  const float* wih = (const float*)d_in[2];
  const float* whh = (const float*)d_in[3];
  const float* bih = (const float*)d_in[4];
  const float* bhh = (const float*)d_in[5];
  float* out = (float*)d_out;                            // (1,S,B,H)
  float* hn  = out + (size_t)S_LEN * BATCH * HDIM;       // (1,B,H)

  char* ws = (char*)d_ws;
  size_t off = 0;
  auto alloc = [&](size_t bytes) {
    void* p = ws + off;
    off = (off + bytes + 255) & ~(size_t)255;
    return p;
  };
  unsigned* bar   = (unsigned*)alloc(256);
  float* biaszn   = (float*)alloc((size_t)GDIM * sizeof(float));
  _Float16* wih16 = (_Float16*)alloc((size_t)GDIM * IDIM * sizeof(_Float16));
  _Float16* whh16 = (_Float16*)alloc((size_t)GDIM * HDIM * sizeof(_Float16));
  float* hbuf     = (float*)alloc((size_t)2 * BATCH * HDIM * sizeof(float));
  _Float16* h16   = (_Float16*)alloc((size_t)2 * BATCH * HDIM * sizeof(_Float16));
  float* gxbuf    = (float*)alloc((size_t)S_LEN * BATCH * GDIM * sizeof(float));
  (void)ws_size; (void)in_sizes; (void)n_in; (void)out_size;

  hipMemsetAsync(bar, 0, 8, stream);  // graph-capturable memset node
  gru_prep<<<512, 256, 0, stream>>>(wih, whh, bih, bhh, wih16, whh16, biaszn);
  gru_gx<<<16384, 256, 0, stream>>>(x, wih16, biaszn, gxbuf);
  gru_recur<<<16, 256, 0, stream>>>(gxbuf, h0, whh16, out, hn, hbuf, h16, bar, 16);
}
